// NeighborAttention_1279900254901
// MI455X (gfx1250) — matrix operand order repaired and checked by reading, not compiled, not run on hardware
//
#include <hip/hip_runtime.h>
#include <hip/hip_bf16.h>
#include <cstdint>

typedef __attribute__((ext_vector_type(16))) _Float16 v16h;
typedef __attribute__((ext_vector_type(8)))  float    v8f;

// ---------------------------------------------------------------------------
// Pack a row-major f32 weight [fin, fout_real] into f16 WMMA B-fragment order,
// kt-major so each K-panel (one kt, all nt) is contiguous:
//   Wp[(kt*nNt + nt)*512 + lane*16 + i]
// Order used here (dense 16-bit B 32x16): lane holds column n = lane%16;
// lanes 0-15 hold K=0..15, lanes 16-31 hold K=16..31 (contiguous).
// Columns >= fout_real are zero-padded (used for the 4-wide logit head).
// ---------------------------------------------------------------------------
__global__ __launch_bounds__(32)
void pack_weight_k(const float* __restrict__ W, _Float16* __restrict__ Wp,
                   int fin, int fout_real, int nNt) {
  int tile = blockIdx.x;           // kt*nNt + nt
  int kt   = tile / nNt;
  int nt   = tile % nNt;
  int lane = threadIdx.x;
  int half = lane >> 4;
  int n    = nt * 16 + (lane & 15);
  _Float16* dst = Wp + (size_t)tile * 512 + (size_t)lane * 16;
#pragma unroll
  for (int i = 0; i < 16; ++i) {
    int k = kt * 32 + half * 16 + i;
    float v = (n < fout_real) ? W[(size_t)k * fout_real + n] : 0.0f;
    dst[i] = (_Float16)v;
  }
  (void)fin;
}

// ---------------------------------------------------------------------------
// 256 threads = 8 waves; wave w computes row-tile (blockIdx.x*8 + w), all
// NNT*16 output columns, with v_wmma_f32_16x16x32_f16.
// Weights: K-panel (NNT tiles, 512 f16 each) staged in LDS, double-buffered.
// Pipeline per K-step: issue A loads -> issue next-panel global loads (stay in
// flight during compute since loadcnt retires in order) -> convert A -> 16x
// wmma reading B fragments from LDS -> ds_store next panel -> one barrier.
// A fragment (16-bit A 16x32 layout): lane row M = lane%16; lanes 0-15 hold
// K = k0..k0+15 with k0 = kt*32 (+16 for lanes 16-31), matching B packing.
// AMODE 0: A = dense [rows, FIN].  AMODE 1: A = concat(h_V[center], h_E).
// ACT: 0 none, 1 ReLU, 2 exact GELU.  scale applied after bias, before act.
// ---------------------------------------------------------------------------
template<int FIN, int NNT, int AMODE, int ACT>
__global__ __launch_bounds__(256)
void gemm_wmma_k(const float* __restrict__ A,
                 const float* __restrict__ hV,
                 const float* __restrict__ hE,
                 const int*   __restrict__ center,
                 const _Float16* __restrict__ Wp,
                 const float* __restrict__ bias, int bias_n,
                 float scale,
                 float* __restrict__ C, int c_stride, int c_cols, int rows) {
  constexpr int NKT     = FIN / 32;
  constexpr int PANEL_H = NNT * 512;       // f16 elements per K-panel
  constexpr int PANEL_B = PANEL_H * 2;     // bytes per K-panel
  constexpr int NSTG    = (PANEL_B + 4095) / 4096;  // 16B chunks per thread

  __shared__ _Float16 smem[2][PANEL_H];

  const int tid   = threadIdx.x;
  const int wid   = tid >> 5;
  const int lane  = tid & 31;
  const int half  = lane >> 4;
  const int mlane = lane & 15;

  const int mtile = blockIdx.x * 8 + wid;
  const int row   = mtile * 16 + mlane;
  const int lrow  = (row < rows) ? row : (rows - 1);  // clamp ragged edge

  int cid = 0;
  if (AMODE == 1) cid = center[lrow];

  v8f  acc[NNT] = {};
  uint4 stage[NSTG];

  // ---- prologue: panel 0 -> LDS buffer 0 ----
#pragma unroll
  for (int s = 0; s < NSTG; ++s) {
    int off = tid * 16 + s * 4096;
    if (off < PANEL_B)
      stage[s] = *reinterpret_cast<const uint4*>((const char*)Wp + off);
  }
#pragma unroll
  for (int s = 0; s < NSTG; ++s) {
    int off = tid * 16 + s * 4096;
    if (off < PANEL_B)
      *reinterpret_cast<uint4*>((char*)&smem[0][0] + off) = stage[s];
  }
  __syncthreads();

  for (int kt = 0; kt < NKT; ++kt) {
    // A-fragment loads first: the wait on them leaves the panel prefetch
    // (issued below) still outstanding during the wmma burst.
    const int k0 = kt * 32 + half * 16;
    float af[16];
#pragma unroll
    for (int seg = 0; seg < 2; ++seg) {
      const int kk = k0 + seg * 8;
      const float* src;
      if (AMODE == 0) {
        src = A + (size_t)lrow * FIN + kk;
      } else {
        // concat(h_V[center] [0..255], h_E [256..767]); 8-wide runs start on
        // multiples of 8, so a run never straddles the 256 seam.
        src = (kk < 256) ? (hV + (size_t)cid * 256 + kk)
                         : (hE + (size_t)lrow * 512 + (kk - 256));
      }
#pragma unroll
      for (int i = 0; i < 8; ++i) af[seg * 8 + i] = src[i];
    }

    // prefetch next K-panel into registers (in flight during compute)
    if (kt + 1 < NKT) {
#pragma unroll
      for (int s = 0; s < NSTG; ++s) {
        int off = tid * 16 + s * 4096;
        if (off < PANEL_B)
          stage[s] = *reinterpret_cast<const uint4*>(
              (const char*)Wp + (size_t)(kt + 1) * PANEL_B + off);
      }
    }

    v16h a;
#pragma unroll
    for (int i = 0; i < 16; ++i) a[i] = (_Float16)af[i];

    const _Float16* bpan = &smem[kt & 1][0];
#pragma unroll
    for (int nt = 0; nt < NNT; ++nt) {
      const v16h b = *reinterpret_cast<const v16h*>(bpan + nt * 512 + lane * 16);
      acc[nt] = __builtin_amdgcn_wmma_f32_16x16x32_f16(
          false, a, false, b, (short)0, acc[nt], false, false);
    }

    // commit next panel to the other LDS buffer; one barrier per step is
    // sufficient for the 2-buffer scheme.
    if (kt + 1 < NKT) {
      char* dstb = (char*)&smem[(kt + 1) & 1][0];
#pragma unroll
      for (int s = 0; s < NSTG; ++s) {
        int off = tid * 16 + s * 4096;
        if (off < PANEL_B)
          *reinterpret_cast<uint4*>(dstb + off) = stage[s];
      }
    }
    __syncthreads();
  }

  // Epilogue. D layout: VGPR r -> M = half*8 + r, N = lane%16 (per 16x16 tile)
#pragma unroll
  for (int nt = 0; nt < NNT; ++nt) {
    const int n  = nt * 16 + mlane;
    const float bv = (n < bias_n) ? bias[n] : 0.0f;
#pragma unroll
    for (int r = 0; r < 8; ++r) {
      const int m = mtile * 16 + half * 8 + r;
      if (m < rows && n < c_cols) {
        float x = (acc[nt][r] + bv) * scale;
        if (ACT == 1)      x = (x > 0.0f) ? x : 0.0f;
        else if (ACT == 2) x = 0.5f * x * (1.0f + erff(x * 0.70710678118654752f));
        C[(size_t)m * c_stride + n] = x;
      }
    }
  }
}

// ---------------------------------------------------------------------------
// Scatter softmax helpers: order-preserving float <-> uint monotone mapping so
// segment-max can use native GLOBAL_ATOMIC_MAX_U32.
// ---------------------------------------------------------------------------
__device__ __forceinline__ unsigned f2mono(float f) {
  unsigned u = __float_as_uint(f);
  return (u & 0x80000000u) ? ~u : (u | 0x80000000u);
}
__device__ __forceinline__ float mono2f(unsigned u) {
  return (u & 0x80000000u) ? __uint_as_float(u & 0x7fffffffu)
                           : __uint_as_float(~u);
}

__global__ void fill_zero_k(float* __restrict__ p, long long n) {
  long long i = (long long)blockIdx.x * blockDim.x + threadIdx.x;
  if (i < n) p[i] = 0.0f;
}

__global__ void seg_max_k(const float* __restrict__ logits,
                          const int* __restrict__ center,
                          unsigned* __restrict__ nmax, int E) {
  int i = blockIdx.x * blockDim.x + threadIdx.x;
  if (i >= E * 4) return;
  int e = i >> 2, h = i & 3;
  atomicMax(&nmax[(size_t)center[e] * 4 + h], f2mono(logits[i]));
}

__global__ void seg_exp_k(const float* __restrict__ logits,
                          const int* __restrict__ center,
                          const unsigned* __restrict__ nmax,
                          float* __restrict__ pbuf,
                          float* __restrict__ nsum, int E) {
  int i = blockIdx.x * blockDim.x + threadIdx.x;
  if (i >= E * 4) return;
  int e = i >> 2, h = i & 3;
  int cid = center[e];
  float p = __expf(logits[i] - mono2f(nmax[(size_t)cid * 4 + h]));
  pbuf[i] = p;
  atomicAdd(&nsum[(size_t)cid * 4 + h], p);
}

// attend[e,h] * v[e, h*64+c'] scatter-added into out_pre[center[e], :]
__global__ void scatter_v_k(const float* __restrict__ pbuf,
                            const float* __restrict__ nsum,
                            const float* __restrict__ v,
                            const int* __restrict__ center,
                            float* __restrict__ outpre, int E) {
  int i = blockIdx.x * blockDim.x + threadIdx.x;   // E*256 threads
  if (i >= E * 256) return;
  int e = i >> 8, c = i & 255, h = c >> 6;
  int cid = center[e];
  float a = pbuf[(size_t)e * 4 + h] / nsum[(size_t)cid * 4 + h];
  atomicAdd(&outpre[(size_t)cid * 256 + c], a * v[(size_t)e * 256 + c]);
}

// ---------------------------------------------------------------------------
extern "C" void kernel_launch(void* const* d_in, const int* in_sizes, int n_in,
                              void* d_out, int out_size, void* d_ws, size_t ws_size,
                              hipStream_t stream) {
  const float* h_V   = (const float*)d_in[0];
  const float* h_E   = (const float*)d_in[1];
  const float* wv1_w = (const float*)d_in[2];
  const float* wv1_b = (const float*)d_in[3];
  const float* wv2_w = (const float*)d_in[4];
  const float* wv2_b = (const float*)d_in[5];
  const float* wv3_w = (const float*)d_in[6];
  const float* wv3_b = (const float*)d_in[7];
  const float* b1_w  = (const float*)d_in[8];
  const float* b1_b  = (const float*)d_in[9];
  const float* b2_w  = (const float*)d_in[10];
  const float* b2_b  = (const float*)d_in[11];
  const float* b3_w  = (const float*)d_in[12];
  const float* b3_b  = (const float*)d_in[13];
  const float* wo_w  = (const float*)d_in[14];
  const int*   center = (const int*)d_in[15];
  float* out = (float*)d_out;

  const int N = in_sizes[0] / 256;
  const int E = in_sizes[15];

  // ---- workspace carve (all 256B aligned) ----
  size_t off = 0;
  auto carve = [&](size_t bytes) -> char* {
    char* p = (char*)d_ws + off;
    off += (bytes + 255) & ~(size_t)255;
    return p;
  };
  float*     bufA   = (float*)carve((size_t)E * 256 * 4);
  float*     bufB   = (float*)carve((size_t)E * 256 * 4);
  float*     logits = (float*)carve((size_t)E * 4 * 4);
  float*     pbuf   = (float*)carve((size_t)E * 4 * 4);
  unsigned*  nmax   = (unsigned*)carve((size_t)N * 4 * 4);   // contiguous with
  float*     nsum   = (float*)carve((size_t)N * 4 * 4);      // nsum, outpre ->
  float*     outpre = (float*)carve((size_t)N * 256 * 4);    // single zero fill
  _Float16*  Wp_b1  = (_Float16*)carve((size_t)(768/32)*(256/16)*512*2);
  _Float16*  Wp_b2  = (_Float16*)carve((size_t)(256/32)*(256/16)*512*2);
  _Float16*  Wp_b3  = (_Float16*)carve((size_t)(256/32)*1*512*2);
  _Float16*  Wp_v1  = (_Float16*)carve((size_t)(512/32)*(256/16)*512*2);
  _Float16*  Wp_v2  = (_Float16*)carve((size_t)(256/32)*(256/16)*512*2);
  _Float16*  Wp_v3  = (_Float16*)carve((size_t)(256/32)*(256/16)*512*2);
  _Float16*  Wp_o   = (_Float16*)carve((size_t)(256/32)*(256/16)*512*2);
  (void)ws_size; (void)n_in; (void)out_size;

  // ---- pack weights to f16 WMMA fragment order (kt-major panels) ----
  pack_weight_k<<<(768/32)*(256/16), 32, 0, stream>>>(b1_w,  Wp_b1, 768, 256, 16);
  pack_weight_k<<<(256/32)*(256/16), 32, 0, stream>>>(b2_w,  Wp_b2, 256, 256, 16);
  pack_weight_k<<<(256/32)*1,        32, 0, stream>>>(b3_w,  Wp_b3, 256, 4,   1);
  pack_weight_k<<<(512/32)*(256/16), 32, 0, stream>>>(wv1_w, Wp_v1, 512, 256, 16);
  pack_weight_k<<<(256/32)*(256/16), 32, 0, stream>>>(wv2_w, Wp_v2, 256, 256, 16);
  pack_weight_k<<<(256/32)*(256/16), 32, 0, stream>>>(wv3_w, Wp_v3, 256, 256, 16);
  pack_weight_k<<<(256/32)*(256/16), 32, 0, stream>>>(wo_w,  Wp_o,  256, 256, 16);

  // ---- zero node accumulators (nmax | nsum | outpre are contiguous) ----
  {
    long long zn = (long long)N * (4 + 4 + 256);
    fill_zero_k<<<(unsigned)((zn + 255) / 256), 256, 0, stream>>>((float*)nmax, zn);
  }

  const int gE = (E + 127) / 128;   // 8 row-tiles of 16 per block
  const int gN = (N + 127) / 128;

  // ---- bias MLP: x1 = ReLU(concat(h_V[c], h_E) @ b1 + b1_b) ----
  gemm_wmma_k<768,16,1,1><<<gE, 256, 0, stream>>>(
      nullptr, h_V, h_E, center, Wp_b1, b1_b, 256, 1.0f, bufA, 256, 256, E);
  // x2 = ReLU(x1 @ b2 + b2_b)
  gemm_wmma_k<256,16,0,1><<<gE, 256, 0, stream>>>(
      bufA, nullptr, nullptr, nullptr, Wp_b2, b2_b, 256, 1.0f, bufB, 256, 256, E);
  // logits = (x2 @ b3 + b3_b) / sqrt(64)
  gemm_wmma_k<256,1,0,0><<<gE, 256, 0, stream>>>(
      bufB, nullptr, nullptr, nullptr, Wp_b3, b3_b, 4, 0.125f, logits, 4, 4, E);

  // ---- value MLP: v1 = GELU(h_E @ wv1 + b)  (bufA reusable after x2) ----
  gemm_wmma_k<512,16,0,2><<<gE, 256, 0, stream>>>(
      h_E, nullptr, nullptr, nullptr, Wp_v1, wv1_b, 256, 1.0f, bufA, 256, 256, E);
  // v2 = GELU(v1 @ wv2 + b)   (bufB reusable after logits)
  gemm_wmma_k<256,16,0,2><<<gE, 256, 0, stream>>>(
      bufA, nullptr, nullptr, nullptr, Wp_v2, wv2_b, 256, 1.0f, bufB, 256, 256, E);
  // v = v2 @ wv3 + b
  gemm_wmma_k<256,16,0,0><<<gE, 256, 0, stream>>>(
      bufB, nullptr, nullptr, nullptr, Wp_v3, wv3_b, 256, 1.0f, bufA, 256, 256, E);

  // ---- scatter softmax over edges sharing a center node ----
  {
    int t = E * 4;
    seg_max_k<<<(t + 255) / 256, 256, 0, stream>>>(logits, center, nmax, E);
    seg_exp_k<<<(t + 255) / 256, 256, 0, stream>>>(logits, center, nmax, pbuf, nsum, E);
  }
  // weighted scatter-sum of values into out_pre[N,256]
  scatter_v_k<<<E, 256, 0, stream>>>(pbuf, nsum, bufA, center, outpre, E);

  // ---- final projection: out = out_pre @ wo_w (no bias) ----
  gemm_wmma_k<256,16,0,0><<<gN, 256, 0, stream>>>(
      outpre, nullptr, nullptr, nullptr, Wp_o, nullptr, 0, 1.0f, out, 256, 256, N);
}
